// Decoder_35416300323555
// MI455X (gfx1250) — compile-verified, matrix operand order checked by reading, not run on hardware
//
#include <hip/hip_runtime.h>

// ---------------- problem constants ----------------
#define LAYERS 6
#define HEADS  8
#define DMODEL 512
#define DHEAD  64
#define DFF    2048
#define BATCH  16
#define SEQ    512
#define BS     (BATCH * SEQ)          // 8192 rows
#define EPS    1e-5f

typedef __bf16 bf16;
typedef __attribute__((ext_vector_type(8)))  bf16  v8bf;
typedef __attribute__((ext_vector_type(16))) bf16  v16bf;
typedef __attribute__((ext_vector_type(8)))  float v8f;

// Load a 16-bit WMMA operand fragment: 8 elems at p, 8 elems at p+16
// (matches CDNA5 16-bit A/B VGPR layout when the caller picks the lane base).
static __device__ __forceinline__ v16bf frag16(const bf16* p) {
    v8bf lo = *(const v8bf*)(p);
    v8bf hi = *(const v8bf*)(p + 16);
    return __builtin_shufflevector(lo, hi, 0,1,2,3,4,5,6,7,8,9,10,11,12,13,14,15);
}

static __device__ __forceinline__ v8f wmma_bf16(v16bf a, v16bf b, v8f c) {
    return __builtin_amdgcn_wmma_f32_16x16x32_bf16(false, a, false, b, (short)0, c, false, false);
}

// ---- CDNA5 async global->LDS copy (ASYNCcnt-tracked), 16 bytes per lane ----
// VDST VGPR holds the LDS byte address ([31:0] of the generic pointer), VADDR
// holds the 64-bit global address (GV mode, SADDR=off).
static __device__ __forceinline__ void async_copy_b128(unsigned lds_off, const void* gaddr) {
    asm volatile("global_load_async_to_lds_b128 %0, %1, off"
                 :: "v"(lds_off), "v"(gaddr) : "memory");
}
static __device__ __forceinline__ void wait_async0() {
    asm volatile("s_wait_asynccnt 0x0" ::: "memory");
}

// =======================================================================
// GEMM: C[M,N] (f32) = A[M,K] (bf16, row-major) * Bt[N,K]^T (bf16, N-major)
// 256 threads = 8 waves; block tile 128x128; K-chunk 32; wave tile 64x32.
// Double-buffered LDS fed by async-to-LDS DMA; one barrier per K-chunk.
// =======================================================================
__global__ __launch_bounds__(256)
void gemm_bf16_nt(const bf16* __restrict__ A, const bf16* __restrict__ Bt,
                  float* __restrict__ C, int M, int N, int K) {
    __shared__ __align__(16) bf16 As[2][128 * 40];
    __shared__ __align__(16) bf16 Bs[2][128 * 40];

    const int tid  = threadIdx.x;
    const int lane = tid & 31;
    const int w    = tid >> 5;
    const int wm   = w >> 2;          // 0..1  (64-row group)
    const int wn   = w & 3;           // 0..3  (32-col group)
    const int half = lane >> 4;
    const int lr   = lane & 15;
    const int row0 = blockIdx.y * 128;
    const int col0 = blockIdx.x * 128;
    const int kOff = half ? 8 : 0;

    // this thread's two 16B staging chunks (same for A and B tiles)
    const int r0 = tid >> 2,          cc0 = (tid & 3) * 8;
    const int r1 = (tid + 256) >> 2,  cc1 = ((tid + 256) & 3) * 8;

    v8f acc[4][2];
    v8f zero = {};
#pragma unroll
    for (int mi = 0; mi < 4; ++mi)
#pragma unroll
        for (int ni = 0; ni < 2; ++ni) acc[mi][ni] = zero;

    auto issue = [&](int buf, int k0) {
        async_copy_b128((unsigned)(uintptr_t)&As[buf][r0 * 40 + cc0],
                        &A [(size_t)(row0 + r0) * K + k0 + cc0]);
        async_copy_b128((unsigned)(uintptr_t)&As[buf][r1 * 40 + cc1],
                        &A [(size_t)(row0 + r1) * K + k0 + cc1]);
        async_copy_b128((unsigned)(uintptr_t)&Bs[buf][r0 * 40 + cc0],
                        &Bt[(size_t)(col0 + r0) * K + k0 + cc0]);
        async_copy_b128((unsigned)(uintptr_t)&Bs[buf][r1 * 40 + cc1],
                        &Bt[(size_t)(col0 + r1) * K + k0 + cc1]);
    };

    const int nk = K >> 5;
    issue(0, 0);
    for (int i = 0; i < nk; ++i) {
        const int buf = i & 1;
        wait_async0();        // my slab-i transfers have landed in LDS
        __syncthreads();      // everyone's landed; buf^1 no longer being read
        if (i + 1 < nk) issue(buf ^ 1, (i + 1) << 5);
        if (i + 2 < nk) {     // cache hint two slabs ahead
            __builtin_prefetch(&A [(size_t)(row0 + (tid >> 1)) * K + ((i + 2) << 5)], 0, 1);
            __builtin_prefetch(&Bt[(size_t)(col0 + (tid >> 1)) * K + ((i + 2) << 5)], 0, 1);
        }

        v16bf af[4], bfr[2];
#pragma unroll
        for (int mi = 0; mi < 4; ++mi)
            af[mi] = frag16(&As[buf][(wm * 64 + mi * 16 + lr) * 40 + kOff]);
#pragma unroll
        for (int ni = 0; ni < 2; ++ni)
            bfr[ni] = frag16(&Bs[buf][(wn * 32 + ni * 16 + lr) * 40 + kOff]);

#pragma unroll
        for (int mi = 0; mi < 4; ++mi)
#pragma unroll
            for (int ni = 0; ni < 2; ++ni)
                acc[mi][ni] = wmma_bf16(af[mi], bfr[ni], acc[mi][ni]);
    }

#pragma unroll
    for (int mi = 0; mi < 4; ++mi)
#pragma unroll
        for (int ni = 0; ni < 2; ++ni) {
            int r = row0 + wm * 64 + mi * 16 + 8 * half;
            int c = col0 + wn * 32 + ni * 16 + lr;
#pragma unroll
            for (int j = 0; j < 8; ++j)
                C[(size_t)(r + j) * N + c] = acc[mi][ni][j];
        }
}

// =======================================================================
// Flash attention (bf16 in, bf16 out, fp32 online softmax).
// grid = (B*H, SEQ/64); block = 128 (4 waves, 16 query rows per wave).
// Q,K: [B*S, 512] row-major with head offset h*64. Vt: [(b*H+h)*64+dh][T].
// =======================================================================
__global__ __launch_bounds__(128)
void flash_attn(const bf16* __restrict__ Q, const bf16* __restrict__ Kb,
                const bf16* __restrict__ Vt, bf16* __restrict__ AO,
                int Tk, int causal) {
    const int bh   = blockIdx.x;
    const int b    = bh >> 3;
    const int h    = bh & 7;
    const int w    = threadIdx.x >> 5;
    const int lane = threadIdx.x & 31;
    const int half = lane >> 4;
    const int lr   = lane & 15;
    const int q0   = blockIdx.y * 64 + w * 16;
    const int kOff = half ? 8 : 0;

    const bf16* Qbase = Q  + (size_t)(b * SEQ) * DMODEL + h * DHEAD;
    const bf16* Kbase = Kb + (size_t)(b * Tk)  * DMODEL + h * DHEAD;
    const bf16* Vbase = Vt + (size_t)(bh * DHEAD) * Tk;

    v16bf qf[2];
#pragma unroll
    for (int c = 0; c < 2; ++c)
        qf[c] = frag16(Qbase + (size_t)(q0 + lr) * DMODEL + c * 32 + kOff);

    v8f o[4];
    v8f zero = {};
#pragma unroll
    for (int d = 0; d < 4; ++d) o[d] = zero;
    float rmax[8], rsum[8];
#pragma unroll
    for (int j = 0; j < 8; ++j) { rmax[j] = -1e30f; rsum[j] = 0.f; }

    __shared__ __align__(16) bf16 pbuf[4][16 * 40];
    bf16* pb = &pbuf[w][0];

    const int tmax = causal ? (q0 + 16) : Tk;
    for (int t0 = 0; t0 < tmax; t0 += 32) {
        // ---- scores: 2 n-tiles x (DH=64 as two 32-K chunks) ----
        v8f sc[2];
#pragma unroll
        for (int nt = 0; nt < 2; ++nt) {
            const bf16* kp = Kbase + (size_t)(t0 + nt * 16 + lr) * DMODEL;
            v16bf k0f = frag16(kp + kOff);
            v16bf k1f = frag16(kp + 32 + kOff);
            sc[nt] = wmma_bf16(qf[0], k0f, zero);
            sc[nt] = wmma_bf16(qf[1], k1f, sc[nt]);
        }
        // ---- scale + causal mask ----
#pragma unroll
        for (int nt = 0; nt < 2; ++nt)
#pragma unroll
            for (int j = 0; j < 8; ++j) {
                float s    = sc[nt][j] * 0.125f;          // 1/sqrt(64)
                int   tcol = t0 + nt * 16 + lr;
                int   rowg = q0 + j + 8 * half;
                if (causal && tcol > rowg) s = -1e30f;
                sc[nt][j] = s;
            }
        // ---- online softmax (row stats per accumulator register) ----
#pragma unroll
        for (int j = 0; j < 8; ++j) {
            float m = fmaxf(sc[0][j], sc[1][j]);
            m = fmaxf(m, __shfl_xor(m, 1, 32));
            m = fmaxf(m, __shfl_xor(m, 2, 32));
            m = fmaxf(m, __shfl_xor(m, 4, 32));
            m = fmaxf(m, __shfl_xor(m, 8, 32));
            float nm   = fmaxf(rmax[j], m);
            float corr = __expf(rmax[j] - nm);
            float p0   = __expf(sc[0][j] - nm);
            float p1   = __expf(sc[1][j] - nm);
            float ps   = p0 + p1;
            ps += __shfl_xor(ps, 1, 32);
            ps += __shfl_xor(ps, 2, 32);
            ps += __shfl_xor(ps, 4, 32);
            ps += __shfl_xor(ps, 8, 32);
            rsum[j] = rsum[j] * corr + ps;
            rmax[j] = nm;
            sc[0][j] = p0; sc[1][j] = p1;
#pragma unroll
            for (int d = 0; d < 4; ++d) o[d][j] *= corr;
        }
        // ---- P: C-layout -> A-layout via per-wave LDS bounce ----
#pragma unroll
        for (int nt = 0; nt < 2; ++nt)
#pragma unroll
            for (int j = 0; j < 8; ++j)
                pb[(j + 8 * half) * 40 + nt * 16 + lr] = (bf16)sc[nt][j];
        __builtin_amdgcn_wave_barrier();   // LDS is in-order within a wave
        v16bf pf = frag16(&pb[lr * 40 + kOff]);
        // ---- PV: 4 dh-tiles ----
#pragma unroll
        for (int d = 0; d < 4; ++d) {
            const bf16* vp = Vbase + (size_t)(d * 16 + lr) * Tk + t0 + kOff;
            o[d] = wmma_bf16(pf, frag16(vp), o[d]);
        }
        __builtin_amdgcn_wave_barrier();
    }

    bf16* obase = AO + (size_t)(b * SEQ) * DMODEL + h * DHEAD;
#pragma unroll
    for (int j = 0; j < 8; ++j) {
        float inv = 1.f / rsum[j];
#pragma unroll
        for (int d = 0; d < 4; ++d)
            obase[(size_t)(q0 + j + 8 * half) * DMODEL + d * 16 + lr] =
                (bf16)(o[d][j] * inv);
    }
}

// =======================================================================
// LayerNorm( a + res [+ bias] ) -> f32 out + bf16 out. One row per block.
// =======================================================================
__global__ __launch_bounds__(128)
void add_ln_k(const float* __restrict__ a, const float* __restrict__ r,
              const float* __restrict__ bias, int hasBias,
              const float* __restrict__ g, const float* __restrict__ be,
              float* __restrict__ outf, bf16* __restrict__ outb) {
    const int row = blockIdx.x;
    const int tid = threadIdx.x;
    const int col = tid * 4;
    float4 va = *(const float4*)(a + (size_t)row * DMODEL + col);
    float4 vr = *(const float4*)(r + (size_t)row * DMODEL + col);
    float x[4] = {va.x + vr.x, va.y + vr.y, va.z + vr.z, va.w + vr.w};
    if (hasBias) {
        float4 vb = *(const float4*)(bias + col);
        x[0] += vb.x; x[1] += vb.y; x[2] += vb.z; x[3] += vb.w;
    }
    float s = 0.f, s2 = 0.f;
#pragma unroll
    for (int j = 0; j < 4; ++j) { s += x[j]; s2 += x[j] * x[j]; }
#pragma unroll
    for (int m = 1; m < 32; m <<= 1) {
        s  += __shfl_xor(s,  m, 32);
        s2 += __shfl_xor(s2, m, 32);
    }
    __shared__ float ps[4], ps2[4];
    int w = tid >> 5, lane = tid & 31;
    if (lane == 0) { ps[w] = s; ps2[w] = s2; }
    __syncthreads();
    s  = ps[0]  + ps[1]  + ps[2]  + ps[3];
    s2 = ps2[0] + ps2[1] + ps2[2] + ps2[3];
    float mean = s * (1.f / DMODEL);
    float var  = s2 * (1.f / DMODEL) - mean * mean;
    float rstd = rsqrtf(var + EPS);
    float4 vg = *(const float4*)(g  + col);
    float4 vb = *(const float4*)(be + col);
    float gg[4] = {vg.x, vg.y, vg.z, vg.w};
    float bb[4] = {vb.x, vb.y, vb.z, vb.w};
    float y[4];
#pragma unroll
    for (int j = 0; j < 4; ++j) y[j] = (x[j] - mean) * rstd * gg[j] + bb[j];
    *(float4*)(outf + (size_t)row * DMODEL + col) = make_float4(y[0], y[1], y[2], y[3]);
#pragma unroll
    for (int j = 0; j < 4; ++j) outb[(size_t)row * DMODEL + col + j] = (bf16)y[j];
}

// ----------------------- small elementwise kernels -----------------------
__global__ void to_bf16_k(const float* __restrict__ in, bf16* __restrict__ out, int n) {
    int i = blockIdx.x * blockDim.x + threadIdx.x;
    if (i < n) out[i] = (bf16)in[i];
}

__global__ void bias_relu_k(const float* __restrict__ in, const float* __restrict__ bias,
                            bf16* __restrict__ out, int n, int N) {
    int i = blockIdx.x * blockDim.x + threadIdx.x;
    if (i < n) {
        float v = in[i] + bias[i % N];
        out[i] = (bf16)fmaxf(v, 0.f);
    }
}

// V f32 [b*T+t][h*64+k] -> Vt bf16 [((b*H+h)*64+k)][t]
__global__ void transpose_v_k(const float* __restrict__ V, bf16* __restrict__ Vt) {
    int i = blockIdx.x * blockDim.x + threadIdx.x;
    if (i >= BS * DMODEL) return;
    int row = i >> 9, col = i & 511;
    int b = row >> 9, t = row & 511;
    int h = col >> 6, k = col & 63;
    Vt[((size_t)(b * HEADS + h) * DHEAD + k) * SEQ + t] = (bf16)V[i];
}

// W [H,D,DH] f32 -> out bf16 [n=h*64+k][d]   (N-major packed projection)
__global__ void pack_qkv_k(const float* __restrict__ W, bf16* __restrict__ out) {
    int o = blockIdx.x * blockDim.x + threadIdx.x;
    if (o >= DMODEL * DMODEL) return;
    int n = o >> 9, d = o & 511;
    int h = n >> 6, k = n & 63;
    out[o] = (bf16)W[((size_t)h * DMODEL + d) * DHEAD + k];
}

// W [K,N] f32 -> out bf16 [n][k]
__global__ void pack_t_k(const float* __restrict__ W, bf16* __restrict__ out, int K, int N) {
    int o = blockIdx.x * blockDim.x + threadIdx.x;
    if (o >= K * N) return;
    int n = o / K, k = o % K;
    out[o] = (bf16)W[(size_t)k * N + n];
}

// =======================================================================
// Host orchestration
// =======================================================================
extern "C" void kernel_launch(void* const* d_in, const int* in_sizes, int n_in,
                              void* d_out, int out_size, void* d_ws, size_t ws_size,
                              hipStream_t stream) {
    (void)in_sizes; (void)n_in; (void)out_size; (void)ws_size;

    const float* x_in  = (const float*)d_in[0];
    const float* y_in  = (const float*)d_in[1];
    const float* Wq1   = (const float*)d_in[2];
    const float* Wk1   = (const float*)d_in[3];
    const float* Wv1   = (const float*)d_in[4];
    const float* Wo1   = (const float*)d_in[5];
    const float* Wq2   = (const float*)d_in[6];
    const float* Wk2   = (const float*)d_in[7];
    const float* Wv2   = (const float*)d_in[8];
    const float* Wo2   = (const float*)d_in[9];
    const float* W1    = (const float*)d_in[10];
    const float* bf1   = (const float*)d_in[11];
    const float* W2    = (const float*)d_in[12];
    const float* bf2   = (const float*)d_in[13];
    const float* ln1g  = (const float*)d_in[14];
    const float* ln1b  = (const float*)d_in[15];
    const float* ln2g  = (const float*)d_in[16];
    const float* ln2b  = (const float*)d_in[17];
    const float* ln3g  = (const float*)d_in[18];
    const float* ln3b  = (const float*)d_in[19];

    // ---- workspace carve-up ----
    char* W_ = (char*)d_ws;
    size_t off = 0;
    auto take = [&](size_t bytes) { size_t o = off; off = (off + bytes + 255) & ~(size_t)255; return o; };
    float* cur = (float*)(W_ + take((size_t)BS * DMODEL * 4));
    float* t0f = (float*)(W_ + take((size_t)BS * DMODEL * 4));
    float* h1f = (float*)(W_ + take((size_t)BS * DMODEL * 4));
    float* h2f = (float*)(W_ + take((size_t)BS * DMODEL * 4));
    float* fff = (float*)(W_ + take((size_t)BS * DFF * 4));
    bf16* actb = (bf16*)(W_ + take((size_t)BS * DMODEL * 2));
    bf16* yb   = (bf16*)(W_ + take((size_t)BS * DMODEL * 2));
    bf16* qb   = (bf16*)(W_ + take((size_t)BS * DMODEL * 2));
    bf16* kb   = (bf16*)(W_ + take((size_t)BS * DMODEL * 2));
    bf16* vtb  = (bf16*)(W_ + take((size_t)BS * DMODEL * 2));
    bf16* aob  = (bf16*)(W_ + take((size_t)BS * DMODEL * 2));
    bf16* ffb  = (bf16*)(W_ + take((size_t)BS * DFF * 2));
    const size_t PROJ = (size_t)DMODEL * DMODEL;              // 262144
    const size_t FFW  = (size_t)DMODEL * DFF;                 // 1048576
    const size_t LW   = 8 * PROJ + 2 * FFW;                   // per-layer packed elems
    bf16* wp = (bf16*)(W_ + take((size_t)LAYERS * LW * 2));

    auto gemm = [&](const bf16* A, const bf16* Bt, float* C, int M, int N, int K) {
        dim3 g(N / 128, M / 128);
        gemm_bf16_nt<<<g, 256, 0, stream>>>(A, Bt, C, M, N, K);
    };
    auto tobf = [&](const float* in, bf16* out, int n) {
        to_bf16_k<<<(n + 255) / 256, 256, 0, stream>>>(in, out, n);
    };

    // ---- one-time packs / converts ----
    for (int l = 0; l < LAYERS; ++l) {
        bf16* base = wp + (size_t)l * LW;
        int gq = (int)(PROJ / 256);
        pack_qkv_k<<<gq, 256, 0, stream>>>(Wq1 + (size_t)l * PROJ, base + 0 * PROJ);
        pack_qkv_k<<<gq, 256, 0, stream>>>(Wk1 + (size_t)l * PROJ, base + 1 * PROJ);
        pack_qkv_k<<<gq, 256, 0, stream>>>(Wv1 + (size_t)l * PROJ, base + 2 * PROJ);
        pack_t_k  <<<gq, 256, 0, stream>>>(Wo1 + (size_t)l * PROJ, base + 3 * PROJ, DMODEL, DMODEL);
        pack_qkv_k<<<gq, 256, 0, stream>>>(Wq2 + (size_t)l * PROJ, base + 4 * PROJ);
        pack_qkv_k<<<gq, 256, 0, stream>>>(Wk2 + (size_t)l * PROJ, base + 5 * PROJ);
        pack_qkv_k<<<gq, 256, 0, stream>>>(Wv2 + (size_t)l * PROJ, base + 6 * PROJ);
        pack_t_k  <<<gq, 256, 0, stream>>>(Wo2 + (size_t)l * PROJ, base + 7 * PROJ, DMODEL, DMODEL);
        pack_t_k<<<(int)(FFW / 256), 256, 0, stream>>>(W1 + (size_t)l * FFW, base + 8 * PROJ,       DMODEL, DFF);
        pack_t_k<<<(int)(FFW / 256), 256, 0, stream>>>(W2 + (size_t)l * FFW, base + 8 * PROJ + FFW, DFF, DMODEL);
    }
    hipMemcpyAsync(cur, x_in, (size_t)BS * DMODEL * 4, hipMemcpyDeviceToDevice, stream);
    tobf(x_in, actb, BS * DMODEL);
    tobf(y_in, yb,   BS * DMODEL);

    const int nAct = BS * DMODEL;
    for (int l = 0; l < LAYERS; ++l) {
        bf16* base = wp + (size_t)l * LW;
        bf16 *wq1t = base + 0 * PROJ, *wk1t = base + 1 * PROJ, *wv1t = base + 2 * PROJ, *wo1t = base + 3 * PROJ;
        bf16 *wq2t = base + 4 * PROJ, *wk2t = base + 5 * PROJ, *wv2t = base + 6 * PROJ, *wo2t = base + 7 * PROJ;
        bf16 *w1t  = base + 8 * PROJ, *w2t  = base + 8 * PROJ + FFW;

        // --- masked self-attention ---
        gemm(actb, wq1t, t0f, BS, DMODEL, DMODEL); tobf(t0f, qb, nAct);
        gemm(actb, wk1t, t0f, BS, DMODEL, DMODEL); tobf(t0f, kb, nAct);
        gemm(actb, wv1t, t0f, BS, DMODEL, DMODEL);
        transpose_v_k<<<nAct / 256, 256, 0, stream>>>(t0f, vtb);
        flash_attn<<<dim3(BATCH * HEADS, SEQ / 64), 128, 0, stream>>>(qb, kb, vtb, aob, SEQ, 1);
        gemm(aob, wo1t, t0f, BS, DMODEL, DMODEL);
        add_ln_k<<<BS, 128, 0, stream>>>(t0f, cur, nullptr, 0,
                                         ln1g + l * DMODEL, ln1b + l * DMODEL, h1f, actb);
        // --- cross-attention (K/V from encoder y) ---
        gemm(actb, wq2t, t0f, BS, DMODEL, DMODEL); tobf(t0f, qb, nAct);
        gemm(yb,   wk2t, t0f, BS, DMODEL, DMODEL); tobf(t0f, kb, nAct);
        gemm(yb,   wv2t, t0f, BS, DMODEL, DMODEL);
        transpose_v_k<<<nAct / 256, 256, 0, stream>>>(t0f, vtb);
        flash_attn<<<dim3(BATCH * HEADS, SEQ / 64), 128, 0, stream>>>(qb, kb, vtb, aob, SEQ, 0);
        gemm(aob, wo2t, t0f, BS, DMODEL, DMODEL);
        add_ln_k<<<BS, 128, 0, stream>>>(t0f, h1f, nullptr, 0,
                                         ln2g + l * DMODEL, ln2b + l * DMODEL, h2f, actb);
        // --- FFN ---
        gemm(actb, w1t, fff, BS, DFF, DMODEL);
        bias_relu_k<<<(BS * DFF + 255) / 256, 256, 0, stream>>>(fff, bf1 + l * DFF, ffb, BS * DFF, DFF);
        gemm(ffb, w2t, t0f, BS, DMODEL, DFF);
        add_ln_k<<<BS, 128, 0, stream>>>(t0f, h2f, bf2 + l * DMODEL, 1,
                                         ln3g + l * DMODEL, ln3b + l * DMODEL, cur, actb);
    }
    hipMemcpyAsync(d_out, cur, (size_t)BS * DMODEL * 4, hipMemcpyDeviceToDevice, stream);
}
